// Correlation_86388972191899
// MI455X (gfx1250) — matrix operand order repaired and checked by reading, not compiled, not run on hardware
//
#include <hip/hip_runtime.h>

typedef _Float16 v8h  __attribute__((ext_vector_type(8)));
typedef _Float16 v16h __attribute__((ext_vector_type(16)));
typedef float    v8f  __attribute__((ext_vector_type(8)));

namespace {
constexpr int Bn   = 16;          // batch
constexpr int Cc   = 256;         // channels (K of the GEMM)
constexpr int H    = 64;
constexpr int W    = 96;
constexpr int MAXD = 20;          // pad == max_displacement
constexpr int S2   = 2;           // stride2
constexpr int Dd   = 21;          // displacements per axis
constexpr int WP   = W + 2*MAXD;  // 136 padded width
constexpr int LDK  = Cc + 8;      // LDS row stride in halves (bank-spread pad)
constexpr int NT   = 768;         // 24 waves: one wave per (M-tile, N-tile)
}

__global__ __launch_bounds__(NT, 1)
void corr_wmma_kernel(const float* __restrict__ f1,
                      const float* __restrict__ f2,
                      float* __restrict__ out)
{
    __shared__ _Float16 As[W  * LDK];   // 96*264*2  = 50688 B : f1 row, x-major over k
    __shared__ _Float16 Bs[WP * LDK];   // 136*264*2 = 71808 B : padded f2 row, xp-major over k

    const int tid = threadIdx.x;
    const int wg  = blockIdx.x;
    const int dy  = wg % Dd;
    const int y   = (wg / Dd) % H;
    const int b0  = wg / (Dd * H);

    // ---- zero Bs: covers the +/-20 x-padding and out-of-range y2 rows ----
    {
        uint32_t* bz = (uint32_t*)Bs;
        #pragma unroll 4
        for (int e = tid; e < (WP*LDK)/2; e += NT) bz[e] = 0u;
    }
    __syncthreads();

    // ---- stage f1 row y: As[x][c] = f16(f1[b, c, y, x]) ----
    {
        const size_t base1 = ((size_t)b0 * Cc * H + y) * W;
        for (int e = tid; e < Cc*W; e += NT) {
            const int c = e / W, x = e - c*W;
            As[x*LDK + c] = (_Float16)f1[base1 + (size_t)c*(H*W) + x];
        }
    }
    // ---- stage f2 row y2 = y + 2*dy - 20 (zero if outside [0,H)) ----
    {
        const int y2 = y + S2*dy - MAXD;
        if (y2 >= 0 && y2 < H) {
            const size_t base2 = ((size_t)b0 * Cc * H + y2) * W;
            for (int e = tid; e < Cc*W; e += NT) {
                const int c = e / W, x = e - c*W;
                Bs[(x + MAXD)*LDK + c] = (_Float16)f2[base2 + (size_t)c*(H*W) + x];
            }
        }
    }
    __syncthreads();

    // ---- one wave per tile-job: i = M-tile (x), j = N-tile (xp band) ----
    const int w    = tid >> 5;     // wave id 0..23
    const int lane = tid & 31;
    const int i    = w >> 2;       // 0..5 : x0 = 16*i
    const int j    = w & 3;        // 0..3 : n0 = x0 + 16*j (padded xp coords)
    const int x0   = i * 16;
    const int n0   = x0 + j * 16;
    const int hi   = lane >> 4;    // half-wave selector
    const int ln   = lane & 15;

    const _Float16* arow = &As[(x0 + ln) * LDK];   // A: row M = ln
    const _Float16* brow = &Bs[(n0 + ln) * LDK];   // B: col N = ln (k-major)

    v8f acc = {};
    #pragma unroll
    for (int kk = 0; kk < Cc/32; ++kk) {
        // A 16x32 f16 fragment: lanes 0-15 K=[0..7,16..23], lanes 16-31 K=[8..15,24..31]
        const int ka = kk*32 + hi*8;
        v8h a_lo = *(const v8h*)(arow + ka);
        v8h a_hi = *(const v8h*)(arow + ka + 16);
        v16h a = __builtin_shufflevector(a_lo, a_hi,
                 0,1,2,3,4,5,6,7, 8,9,10,11,12,13,14,15);
        // B 32x16 f16 fragment: lanes 0-15 K=[0..7,16..23], lanes 16-31 K=[8..15,24..31]
        const int kb = kk*32 + hi*8;
        v8h b_lo = *(const v8h*)(brow + kb);
        v8h b_hi = *(const v8h*)(brow + kb + 16);
        v16h b = __builtin_shufflevector(b_lo, b_hi,
                 0,1,2,3,4,5,6,7, 8,9,10,11,12,13,14,15);
        acc = __builtin_amdgcn_wmma_f32_16x16x32_f16(
                  /*neg_a=*/false, a, /*neg_b=*/false, b,
                  /*c_mod=*/(short)0, acc, /*reuse_a=*/false, /*reuse_b=*/false);
    }

    // ---- scatter the even-offset band: off = xp - x = 16*j + n - m ----
    const float scale = 1.0f / (float)Cc;
    #pragma unroll
    for (int r = 0; r < 8; ++r) {
        const int m   = r + hi*8;          // D row (VGPR r layout)
        const int off = j*16 + ln - m;
        if (off >= 0 && off <= 2*MAXD && !(off & 1)) {
            const int dx = off >> 1;
            const int x  = x0 + m;
            out[(((size_t)b0*(Dd*Dd) + (dy*Dd + dx))*H + y)*W + x] = acc[r] * scale;
        }
    }
}

extern "C" void kernel_launch(void* const* d_in, const int* in_sizes, int n_in,
                              void* d_out, int out_size, void* d_ws, size_t ws_size,
                              hipStream_t stream) {
    (void)in_sizes; (void)n_in; (void)out_size; (void)d_ws; (void)ws_size;
    const float* f1 = (const float*)d_in[0];
    const float* f2 = (const float*)d_in[1];
    float* out = (float*)d_out;
    const int nblocks = Bn * H * Dd;   // 16*64*21 = 21504 (b,y,dy) workgroups
    corr_wmma_kernel<<<nblocks, NT, 0, stream>>>(f1, f2, out);
}
